// SimplexLayer_6047313953379
// MI455X (gfx1250) — matrix operand order repaired and checked by reading, not compiled, not run on hardware
//
#include <hip/hip_runtime.h>
#include <hip/hip_bf16.h>
#include <math.h>

typedef __attribute__((ext_vector_type(16))) __bf16 v16bf;
typedef __attribute__((ext_vector_type(8)))  __bf16 v8bf;
typedef __attribute__((ext_vector_type(8)))  float  v8f;
typedef __attribute__((ext_vector_type(4)))  float  v4f;

#define BDIM 4
#define SDIM 4096
#define DDIM 1024
#define EDIM 8
#define MTOT (BDIM*SDIM)        // 16384 tokens
#define KTOT (EDIM*DDIM)        // 8192  folded K'
#define MT   256                 // M tile (rows)
#define NT   128                 // N tile (cols)
#define KT   32
#define LDA  40                  // bf16 row stride in LDS (80B, keeps 16B align, pads banks)
#define NSTAGE (KTOT/KT)         // 256
#define NTHREADS 512             // 16 waves: 8 (M) x 2 (N), wave tile 32x64

static __device__ __forceinline__ v8f vzero8() {
    v8f z;
#pragma unroll
    for (int i = 0; i < 8; ++i) z[i] = 0.0f;
    return z;
}

// One wave-wide async copy: each lane moves 16B global -> LDS. Tracked by ASYNCcnt.
static __device__ __forceinline__ void async_b128(unsigned lds_addr, const void* gptr) {
    asm volatile("global_load_async_to_lds_b128 %0, %1, off"
                 :: "v"(lds_addr), "v"(gptr) : "memory");
}
static __device__ __forceinline__ void wait_async0() {
    asm volatile("s_wait_asynccnt 0x0" ::: "memory");
}

// ---------------- Kernel 1: W_edges f32 -> bf16 ----------------
__global__ void cvt_w_kernel(const float* __restrict__ w, __bf16* __restrict__ o) {
    size_t i = ((size_t)blockIdx.x * 256u + threadIdx.x) * 8u;
    v4f a = *(const v4f*)(w + i);
    v4f b = *(const v4f*)(w + i + 4);
    v8bf r;
#pragma unroll
    for (int k = 0; k < 4; ++k) { r[k] = (__bf16)a[k]; r[k + 4] = (__bf16)b[k]; }
    *(v8bf*)(o + i) = r;
}

// ---------------- Kernel 2: routing softmax + h f32 -> bf16 ----------------
__global__ void route_kernel(const float* __restrict__ h,
                             const float* __restrict__ Wr,
                             const float* __restrict__ br,
                             __bf16* __restrict__ hbf,
                             float* __restrict__ rw) {
    const int t = blockIdx.x;       // token 0..16383
    const int j = threadIdx.x;      // 0..127, each owns 8 d-values
    const size_t base = (size_t)t * DDIM + j * 8;
    v4f h0 = *(const v4f*)(h + base);
    v4f h1 = *(const v4f*)(h + base + 4);
    v8bf hb;
#pragma unroll
    for (int k = 0; k < 4; ++k) { hb[k] = (__bf16)h0[k]; hb[k + 4] = (__bf16)h1[k]; }
    *(v8bf*)(hbf + base) = hb;

    __shared__ float red[EDIM][128];
#pragma unroll
    for (int e = 0; e < EDIM; ++e) {
        const float* wr = Wr + (size_t)e * DDIM + j * 8;
        v4f w0 = *(const v4f*)wr;
        v4f w1 = *(const v4f*)(wr + 4);
        float s = h0[0]*w0[0] + h0[1]*w0[1] + h0[2]*w0[2] + h0[3]*w0[3]
                + h1[0]*w1[0] + h1[1]*w1[1] + h1[2]*w1[2] + h1[3]*w1[3];
        red[e][j] = s;
    }
    __syncthreads();
    for (int st = 64; st > 0; st >>= 1) {
        if (j < st) {
#pragma unroll
            for (int e = 0; e < EDIM; ++e) red[e][j] += red[e][j + st];
        }
        __syncthreads();
    }
    if (j == 0) {
        float lg[EDIM], mx = -3.0e38f;
#pragma unroll
        for (int e = 0; e < EDIM; ++e) { lg[e] = red[e][0] + br[e]; mx = fmaxf(mx, lg[e]); }
        float s = 0.0f;
#pragma unroll
        for (int e = 0; e < EDIM; ++e) { lg[e] = expf(lg[e] - mx); s += lg[e]; }
        float inv = 1.0f / s;
#pragma unroll
        for (int e = 0; e < EDIM; ++e) rw[(size_t)t * EDIM + e] = lg[e] * inv;
    }
}

// ---------------- Kernel 3: fused expert GEMM + rw fold + bias + GELU ----------------
__launch_bounds__(NTHREADS)
__global__ void moe_gemm_kernel(const __bf16* __restrict__ hbf,
                                const __bf16* __restrict__ wbf,
                                const float* __restrict__ rw,
                                const float* __restrict__ bias,
                                float* __restrict__ out) {
    __shared__ __bf16 As[2][MT * LDA];   // 2 x 20 KB
    __shared__ __bf16 Bs[2][NT * LDA];   // 2 x 10 KB
    __shared__ float  rws[MT][EDIM];     // 8 KB

    const int tid  = threadIdx.x;
    const int lane = tid & 31;
    const int wid  = tid >> 5;        // 0..15
    const int wm   = wid >> 1;        // 0..7 -> 32-row strip
    const int wn   = wid & 1;         // 0..1 -> 64-col strip
    const int m0   = blockIdx.x * MT;
    const int n0   = blockIdx.y * NT;
    const bool lo16 = (lane & 16) == 0;

    // preload routing weights for this row tile: 256 rows x 8 experts (2048 f32)
    {
        v4f r = *(const v4f*)(rw + (size_t)m0 * EDIM + tid * 4);
        *(v4f*)(&rws[0][0] + tid * 4) = r;
    }

    auto issue_stage = [&](int s, int buf) {
        const int e  = s >> 5;              // expert segment
        const int kk = (s & 31) * KT;       // d offset within segment
        const unsigned abase = (unsigned)(uintptr_t)(&As[buf][0]);
        const unsigned bbase = (unsigned)(uintptr_t)(&Bs[buf][0]);
#pragma unroll
        for (int i = 0; i < 2; ++i) {       // A tile: 256 x 32 bf16 (1024 chunks)
            const int c   = (tid << 1) + i;
            const int row = c >> 2;
            const int kc  = (c & 3) << 3;
            const __bf16* g = hbf + (size_t)(m0 + row) * DDIM + (kk + kc);
            async_b128(abase + (unsigned)(row * LDA + kc) * 2u, g);
        }
        {                                   // B tile: 128 cols x 32 k (512 chunks)
            const int c   = tid;
            const int col = c >> 2;
            const int kc  = (c & 3) << 3;
            const __bf16* g = wbf + ((size_t)e * DDIM + (n0 + col)) * (size_t)DDIM + (kk + kc);
            async_b128(bbase + (unsigned)(col * LDA + kc) * 2u, g);
        }
    };

    v8f accF[2][4], accE[2][4];
#pragma unroll
    for (int mi = 0; mi < 2; ++mi)
#pragma unroll
        for (int ni = 0; ni < 4; ++ni) { accF[mi][ni] = vzero8(); accE[mi][ni] = vzero8(); }

    issue_stage(0, 0);

    for (int s = 0; s < NSTAGE; ++s) {
        const int buf = s & 1;
        wait_async0();          // stage s resident in LDS (this wave's copies)
        __syncthreads();        // all waves' copies visible; prior reads of buf^1 done
        if (s + 1 < NSTAGE) issue_stage(s + 1, buf ^ 1);

        // A fragments: 16x32 bf16 per m-subtile. lane holds row (lane&15),
        // k-chunks at kbase=(lane>>4)*8 and kbase+16 (two 16B pieces).
        v16bf af[2];
#pragma unroll
        for (int mi = 0; mi < 2; ++mi) {
            const __bf16* ap = &As[buf][(wm * 32 + mi * 16 + (lane & 15)) * LDA + ((lane >> 4) * 8)];
            v8bf alo = *(const v8bf*)ap;
            v8bf ahi = *(const v8bf*)(ap + 16);
            af[mi] = __builtin_shufflevector(alo, ahi, 0,1,2,3,4,5,6,7,8,9,10,11,12,13,14,15);
        }
        // B fragments: 32x16 bf16 per n-subtile. lane holds column (lane&15),
        // k-chunks at kbase=(lane>>4)*8 and kbase+16 (two 16B pieces).
        v16bf bfm[4];
#pragma unroll
        for (int ni = 0; ni < 4; ++ni) {
            const __bf16* bp = &Bs[buf][(wn * 64 + ni * 16 + (lane & 15)) * LDA + ((lane >> 4) * 8)];
            v8bf blo = *(const v8bf*)bp;
            v8bf bhi = *(const v8bf*)(bp + 16);
            bfm[ni] = __builtin_shufflevector(blo, bhi, 0,1,2,3,4,5,6,7,8,9,10,11,12,13,14,15);
        }
#pragma unroll
        for (int mi = 0; mi < 2; ++mi)
#pragma unroll
            for (int ni = 0; ni < 4; ++ni)
                accE[mi][ni] = __builtin_amdgcn_wmma_f32_16x16x32_bf16(
                    false, af[mi], false, bfm[ni], (short)0, accE[mi][ni], false, false);

        // end of expert segment: fold accE into accF with per-row routing weight
        if ((s & 31) == 31) {
            const int e = s >> 5;
#pragma unroll
            for (int mi = 0; mi < 2; ++mi) {
                v8f sc;
#pragma unroll
                for (int v = 0; v < 8; ++v) {
                    const int row = wm * 32 + mi * 16 + v + (lo16 ? 0 : 8);
                    sc[v] = rws[row][e];
                }
#pragma unroll
                for (int ni = 0; ni < 4; ++ni) {
                    accF[mi][ni] += sc * accE[mi][ni];
                    accE[mi][ni] = vzero8();
                }
            }
        }
    }

    // epilogue: + bias, exact GELU, store.
    // C layout: lanes 0-15 -> N=lane, M=v; lanes 16-31 -> N=lane-16, M=v+8.
#pragma unroll
    for (int mi = 0; mi < 2; ++mi) {
        const int rbase = m0 + wm * 32 + mi * 16 + (lo16 ? 0 : 8);
#pragma unroll
        for (int ni = 0; ni < 4; ++ni) {
            const int col = n0 + wn * 64 + ni * 16 + (lane & 15);
            const float bv = bias[col];
#pragma unroll
            for (int v = 0; v < 8; ++v) {
                float x = accF[mi][ni][v] + bv;
                float g = 0.5f * x * (1.0f + erff(x * 0.70710678118654752440f));
                out[(size_t)(rbase + v) * DDIM + col] = g;
            }
        }
    }
}

extern "C" void kernel_launch(void* const* d_in, const int* in_sizes, int n_in,
                              void* d_out, int out_size, void* d_ws, size_t ws_size,
                              hipStream_t stream) {
    const float* h       = (const float*)d_in[0];   // [B,S,D]
    const float* W_route = (const float*)d_in[1];   // [E,D]
    const float* b_route = (const float*)d_in[2];   // [E]
    const float* W_edges = (const float*)d_in[3];   // [E,D,D]
    const float* l_bias  = (const float*)d_in[4];   // [D]
    float* out = (float*)d_out;

    __bf16* hbf = (__bf16*)d_ws;                                   // 32 MiB
    __bf16* wbf = (__bf16*)((char*)d_ws + (32ull << 20));          // 16 MiB
    float*  rwp = (float*)((char*)d_ws + (48ull << 20));           // 512 KiB

    // 1) convert W_edges to bf16
    cvt_w_kernel<<<(EDIM * DDIM * DDIM) / (256 * 8), 256, 0, stream>>>(W_edges, wbf);
    // 2) routing softmax + h -> bf16
    route_kernel<<<MTOT, 128, 0, stream>>>(h, W_route, b_route, hbf, rwp);
    // 3) fused GEMM (M=16384, N=1024, K'=8192) + fold + bias + GELU
    dim3 grid(MTOT / MT, DDIM / NT);
    moe_gemm_kernel<<<grid, NTHREADS, 0, stream>>>(hbf, wbf, rwp, l_bias, out);
}
